// GaussianImage_Cholesky_15917148799694
// MI455X (gfx1250) — matrix operand order repaired and checked by reading, not compiled, not run on hardware
//
#include <hip/hip_runtime.h>
#include <hip/hip_bf16.h>

typedef __attribute__((ext_vector_type(16))) _Float16 v16h;
typedef __attribute__((ext_vector_type(8)))  float    v8f;
typedef __attribute__((ext_vector_type(4)))  float    v4f;

#define GW 256
#define GH 256
#define GN 1024
#define LOG2E 1.4426950408889634f

// ---------------------------------------------------------------------------
// Kernel 1: per-gaussian preprocessing (N=1024, trivial cost).
// SoA outputs so quad ds_load_b128s are directly v_pk-pair aligned:
//   cxA/cyA/naA/nbA/ncA[n], with -log2(e) and the 0.5 folded in:
//   w = exp2(na*dx^2 + nb*dx*dy + nc*dy^2)
// feat4[n*4] = f16 { feat0*op, feat1*op, feat2*op, 0 }  (padded B row)
// ---------------------------------------------------------------------------
__global__ __launch_bounds__(256)
void gs_prep_kernel(const float* __restrict__ xyz,
                    const float* __restrict__ chol,
                    const float* __restrict__ opac,
                    const float* __restrict__ feat,
                    float* __restrict__ cxA, float* __restrict__ cyA,
                    float* __restrict__ naA, float* __restrict__ nbA,
                    float* __restrict__ ncA,
                    _Float16* __restrict__ feat4)
{
    int n = blockIdx.x * 256 + threadIdx.x;
    if (n >= GN) return;

    float mx = tanhf(xyz[2*n + 0]);
    float my = tanhf(xyz[2*n + 1]);
    float cx = 0.5f * GW * (mx + 1.0f);
    float cy = 0.5f * GH * (my + 1.0f);

    float l1 = chol[3*n + 0] + 0.5f;
    float l2 = chol[3*n + 1];
    float l3 = chol[3*n + 2] + 0.5f;
    float a = l1 * l1;
    float b = l1 * l2;
    float c = l2 * l2 + l3 * l3;
    float det = a * c - b * b;
    float inv = 1.0f / det;
    float conic_a =  c * inv;
    float conic_b = -b * inv;
    float conic_c =  a * inv;

    cxA[n] = cx;
    cyA[n] = cy;
    naA[n] = -0.5f * LOG2E * conic_a;
    nbA[n] = -        LOG2E * conic_b;
    ncA[n] = -0.5f * LOG2E * conic_c;

    float o = opac[n];
    feat4[n*4 + 0] = (_Float16)(feat[3*n + 0] * o);
    feat4[n*4 + 1] = (_Float16)(feat[3*n + 1] * o);
    feat4[n*4 + 2] = (_Float16)(feat[3*n + 2] * o);
    feat4[n*4 + 3] = (_Float16)0.0f;
}

// ---------------------------------------------------------------------------
// Kernel 2: splat + WMMA reduction.
// One wave = one 16-pixel x-strip (M tile). K-loop unrolled x2 (64 gaussians
// per iteration, 2 WMMAs), B fragments software-pipelined (each B a fresh
// ds_load destination). Quadratic evaluated in v4f vector ops over SoA LDS
// arrays so every v_pk operand pair comes straight out of a ds_load_b128.
// A (16x32 f16): lane<16 -> row M=lane, K in {0..7,16..23};
//                lane>=16 -> row M=lane-16, K in {8..15,24..31}.
// B (32x16 f16): lane l -> column N=l%16, same K set as A (3 columns used).
// D (16x16 f32): lane%16 = channel, VGPR r -> pixel row r (+8 upper half).
// ---------------------------------------------------------------------------
__global__ __launch_bounds__(256)
void gs_splat_kernel(const float* __restrict__ gparams,  // 5*GN floats (SoA)
                     const _Float16* __restrict__ feat4,
                     float* __restrict__ out)
{
    __shared__ __align__(16) float    scx[GN];     // 4 KB
    __shared__ __align__(16) float    scy[GN];     // 4 KB
    __shared__ __align__(16) float    sna[GN];     // 4 KB
    __shared__ __align__(16) float    snb[GN];     // 4 KB
    __shared__ __align__(16) float    snc[GN];     // 4 KB
    __shared__ __align__(16) _Float16 sf[GN * 4];  // 8 KB: padded B rows

    // Block-cooperative preload of all gaussian data into LDS.
    for (int i = threadIdx.x; i < GN; i += 256) {
        scx[i] = gparams[i];
        scy[i] = gparams[GN + i];
        sna[i] = gparams[2*GN + i];
        snb[i] = gparams[3*GN + i];
        snc[i] = gparams[4*GN + i];
    }
    for (int i = threadIdx.x; i < GN * 4; i += 256) sf[i] = feat4[i];
    __syncthreads();

    const int lane = threadIdx.x & 31;
    const int wave = threadIdx.x >> 5;
    const int tile = blockIdx.x * 8 + wave;   // 0..4095
    const int tx = tile & 15;                 // 16 strips per image row
    const int ty = tile >> 4;                 // image row 0..255
    const int x0 = tx << 4;

    const int   row  = lane & 15;             // pixel row within tile (M)
    const int   koff = (lane >> 4) << 3;      // 0 or 8 (A-layout K offset)
    const float px   = (float)(x0 + row) + 0.5f;
    const float py   = (float)ty + 0.5f;

    // B-column loader: lane l holds column N = l%16 (channel), K = k+koff+{0..7,16..23}.
    auto load_b = [&](int k) {
        v16h r = {};
        const int n = lane & 15;
        if (n < 4) {
#pragma unroll
            for (int s = 0; s < 16; ++s)
                r[s] = sf[(k + koff + (s < 8 ? s : s + 8)) * 4 + n];
        }
        return r;
    };

    // A-fragment compute: 16 exps for gaussians K = k0+koff + {0..7,16..23},
    // processed as quads so all pk operands are load-aligned pairs.
    auto compute_a = [&](int k0) {
        v16h a;
#pragma unroll
        for (int g = 0; g < 2; ++g) {
#pragma unroll
            for (int q = 0; q < 2; ++q) {
                const int kq = k0 + koff + g * 16 + q * 4;
                const v4f cx4 = *(const v4f*)(scx + kq);
                const v4f cy4 = *(const v4f*)(scy + kq);
                const v4f na4 = *(const v4f*)(sna + kq);
                const v4f nb4 = *(const v4f*)(snb + kq);
                const v4f nc4 = *(const v4f*)(snc + kq);
                const v4f dx = px - cx4;
                const v4f dy = py - cy4;
                const v4f e  = na4 * dx * dx + nb4 * dx * dy + nc4 * dy * dy;
#pragma unroll
                for (int t = 0; t < 4; ++t)
                    a[g * 8 + q * 4 + t] =
                        (_Float16)__builtin_amdgcn_exp2f(e[t]);
            }
        }
        return a;
    };

    v8f  acc = {};
    v16h b   = load_b(0);

    for (int k0 = 0; k0 < GN; k0 += 64) {
        // First half: compute A(k0) while prefetching B(k0+32).
        v16h a1 = compute_a(k0);
        v16h b2 = load_b(k0 + 32);
        acc = __builtin_amdgcn_wmma_f32_16x16x32_f16(
            false, a1, false, b, (short)0, acc, false, false);

        // Second half: compute A(k0+32) while prefetching B for next iter.
        v16h a2 = compute_a(k0 + 32);
        b = load_b((k0 + 64) & (GN - 1));
        acc = __builtin_amdgcn_wmma_f32_16x16x32_f16(
            false, a2, false, b2, (short)0, acc, false, false);
    }

    // Store: only channel lanes 0..2 (others hold exact zeros from zero B cols).
    const int chan = lane & 15;
    if (chan < 3) {
        const int mbase = (lane >> 4) << 3;  // 0 or 8
#pragma unroll
        for (int r = 0; r < 8; ++r) {
            float v = acc[r];
            v = v < 0.0f ? 0.0f : (v > 1.0f ? 1.0f : v);
            const int x = x0 + mbase + r;
            out[chan * (GH * GW) + ty * GW + x] = v;
        }
    }
}

// ---------------------------------------------------------------------------
extern "C" void kernel_launch(void* const* d_in, const int* in_sizes, int n_in,
                              void* d_out, int out_size, void* d_ws, size_t ws_size,
                              hipStream_t stream)
{
    const float* xyz  = (const float*)d_in[0];   // (N,2)
    const float* chol = (const float*)d_in[1];   // (N,3)
    const float* opac = (const float*)d_in[2];   // (N,1)
    const float* feat = (const float*)d_in[3];   // (N,3)
    float* out = (float*)d_out;                  // (1,3,H,W) f32

    float* gparams = (float*)d_ws;               // 5*GN floats (SoA) = 20 KB
    float* cxA = gparams;
    float* cyA = gparams + GN;
    float* naA = gparams + 2*GN;
    float* nbA = gparams + 3*GN;
    float* ncA = gparams + 4*GN;
    _Float16* feat4 = (_Float16*)(gparams + 5*GN);  // 8 KB

    gs_prep_kernel<<<(GN + 255) / 256, 256, 0, stream>>>(
        xyz, chol, opac, feat, cxA, cyA, naA, nbA, ncA, feat4);

    // 4096 strips of 16 pixels, 8 waves (256 threads) per block.
    gs_splat_kernel<<<512, 256, 0, stream>>>(gparams, feat4, out);
}
